// NNUE_halfkp_19670950215787
// MI455X (gfx1250) — matrix operand order repaired and checked by reading, not compiled, not run on hardware
//
#include <hip/hip_runtime.h>

typedef __attribute__((ext_vector_type(16))) _Float16 v16h;
typedef __attribute__((ext_vector_type(8)))  _Float16 v8h;
typedef __attribute__((ext_vector_type(8)))  float    v8f;
typedef __attribute__((ext_vector_type(4)))  float    v4f;

#define NFEAT  125388
#define L1     256
#define L2     32
#define L3     32
#define BATCH  1024
#define NF4    (NFEAT / 4)   // 31347 float4 per row
#define MAXIDX 512           // capacity for nonzero list (typ. ~38)

// ---------------------------------------------------------------------------
// Kernel 0: repack W1 [512x32] and W2 [32x32] (f32 row-major) into f16 WMMA
// B-fragment order so the MLP kernel loads one contiguous v16h per lane.
// B order used here (contiguous k, as A): lane l -> col n=l%16; half=l/16;
// vector elem 2j,2j+1 = B[half*16+2j, n], B[half*16+2j+1, n].
// ---------------------------------------------------------------------------
__global__ void __launch_bounds__(256)
repack_weights(const float* __restrict__ W1, const float* __restrict__ W2,
               _Float16* __restrict__ W1f, _Float16* __restrict__ W2f) {
  int id = blockIdx.x * blockDim.x + threadIdx.x;
  const int W1_PAIRS = 16 * 2 * 32 * 8;  // ksteps * ntiles * lanes * pairs
  const int W2_PAIRS = 1 * 2 * 32 * 8;
  if (id < W1_PAIRS) {
    int j    = id & 7;
    int lane = (id >> 3) & 31;
    int nt   = (id >> 8) & 1;
    int ks   = id >> 9;
    int k0   = ks * 32 + (lane >> 4) * 16 + 2 * j;
    int n    = nt * 16 + (lane & 15);
    _Float16* dst = W1f + (((size_t)(ks * 2 + nt) * 32 + lane) * 16 + 2 * j);
    dst[0] = (_Float16)W1[(size_t)k0 * L2 + n];
    dst[1] = (_Float16)W1[(size_t)(k0 + 1) * L2 + n];
  } else if (id < W1_PAIRS + W2_PAIRS) {
    int id2  = id - W1_PAIRS;
    int j    = id2 & 7;
    int lane = (id2 >> 3) & 31;
    int nt   = (id2 >> 8) & 1;
    int k0   = (lane >> 4) * 16 + 2 * j;
    int n    = nt * 16 + (lane & 15);
    _Float16* dst = W2f + (((size_t)nt * 32 + lane) * 16 + 2 * j);
    dst[0] = (_Float16)W2[(size_t)k0 * L3 + n];
    dst[1] = (_Float16)W2[(size_t)(k0 + 1) * L3 + n];
  }
}

// ---------------------------------------------------------------------------
// Kernel 1: feature transformer exploiting sparsity.
// One 256-thread block per (row, perspective).
// Phase 1: stream the dense row with NON-TEMPORAL float4 loads (the 1.03 GB
//   x stream must not evict the 128 MB W_in from the 192 MB L2), 4x unrolled
//   for deeper LOADcnt pipelining; collect nonzero (index,value) into LDS.
// Phase 2: thread t owns output feature t:
//   acc = b_in[t] + sum_i val_i * W_in[idx_i*256 + t]
//   (coalesced 1 KB row gathers, L2-resident; 4 independent partial sums to
//   break the load->FMA dependency chain).
// Clipped-ReLU, stored as f16 into act[row][persp*256 + t].
// ---------------------------------------------------------------------------
__device__ __forceinline__ void scan4(v4f v, int i4, int* s_idx, float* s_val,
                                      int* s_cnt) {
  if (v.x != 0.f || v.y != 0.f || v.z != 0.f || v.w != 0.f) {
#pragma unroll
    for (int c = 0; c < 4; ++c) {
      float f = v[c];
      if (f != 0.f) {
        int pos = atomicAdd(s_cnt, 1);
        if (pos < MAXIDX) { s_idx[pos] = i4 * 4 + c; s_val[pos] = f; }
      }
    }
  }
}

__global__ void __launch_bounds__(256)
feature_transformer(const float* __restrict__ x1, const float* __restrict__ x2,
                    const float* __restrict__ W_in, const float* __restrict__ b_in,
                    _Float16* __restrict__ act) {
  __shared__ int   s_idx[MAXIDX];
  __shared__ float s_val[MAXIDX];
  __shared__ int   s_cnt;
  const int t   = threadIdx.x;
  const int row = blockIdx.x & (BATCH - 1);
  const int p   = blockIdx.x >> 10;
  const float* xrow = (p ? x2 : x1) + (size_t)row * NFEAT;
  if (t == 0) s_cnt = 0;
  __syncthreads();

  const v4f* x4 = (const v4f*)xrow;
  // 4x unrolled non-temporal streaming scan: 4 independent b128 loads/iter.
  for (int i = t; i < NF4; i += 4 * 256) {
    int i1 = i + 256, i2 = i + 512, i3 = i + 768;
    v4f v0 = __builtin_nontemporal_load(&x4[i]);
    v4f v1 = (i1 < NF4) ? __builtin_nontemporal_load(&x4[i1]) : (v4f){};
    v4f v2 = (i2 < NF4) ? __builtin_nontemporal_load(&x4[i2]) : (v4f){};
    v4f v3 = (i3 < NF4) ? __builtin_nontemporal_load(&x4[i3]) : (v4f){};
    scan4(v0, i,  s_idx, s_val, &s_cnt);
    scan4(v1, i1, s_idx, s_val, &s_cnt);
    scan4(v2, i2, s_idx, s_val, &s_cnt);
    scan4(v3, i3, s_idx, s_val, &s_cnt);
  }
  __syncthreads();

  const int cnt = (s_cnt < MAXIDX) ? s_cnt : MAXIDX;
  // 4 independent partial accumulators over the L2-resident W_in gathers.
  float a0 = b_in[t], a1 = 0.f, a2 = 0.f, a3 = 0.f;
  int i = 0;
  for (; i + 4 <= cnt; i += 4) {
    a0 += s_val[i + 0] * W_in[(size_t)s_idx[i + 0] * L1 + t];
    a1 += s_val[i + 1] * W_in[(size_t)s_idx[i + 1] * L1 + t];
    a2 += s_val[i + 2] * W_in[(size_t)s_idx[i + 2] * L1 + t];
    a3 += s_val[i + 3] * W_in[(size_t)s_idx[i + 3] * L1 + t];
  }
  for (; i < cnt; ++i)
    a0 += s_val[i] * W_in[(size_t)s_idx[i] * L1 + t];
  float acc = (a0 + a1) + (a2 + a3);
  acc = fminf(fmaxf(acc, 0.f), 1.f);
  act[(size_t)row * (2 * L1) + (size_t)p * L1 + t] = (_Float16)acc;
}

// ---------------------------------------------------------------------------
// Kernel 2: fused MLP head. One wave (32 threads) per 16-row batch tile.
//  L1: [16x512]f16 @ [512x32]f16 -> f32, +b1, clip   (16 ksteps x 2 ntiles WMMA)
//  L2: [16x32]f16  @ [32x32]f16  -> f32, +b2, clip   (2 WMMA)
//  Out: [16x32]f32 . Wo[32] + bo                     (VALU dot, lanes 0..15)
// A fragments loaded in contiguous k order (as W1f/W2f): lane l -> row l%16,
// 16 f16 at cols (l/16)*16 .. (l/16)*16+15.
// ---------------------------------------------------------------------------
__global__ void __launch_bounds__(32)
mlp_head(const _Float16* __restrict__ act,
         const _Float16* __restrict__ W1f, const float* __restrict__ b1,
         const _Float16* __restrict__ W2f, const float* __restrict__ b2,
         const float* __restrict__ Wo, const float* __restrict__ bo,
         float* __restrict__ out) {
  __shared__ __align__(32) _Float16 t1[16 * 32];
  __shared__ __align__(16) float    t2[16 * 32];
  const int lane = threadIdx.x;
  const int half = lane >> 4;
  const int mloc = lane & 15;
  const int n    = mloc;           // C/D col owned by this lane
  const int m0   = blockIdx.x * 16;

  // ---- Layer 1: act[16x512] @ W1[512x32] ----
  v8f c0 = {}, c1 = {};
  const _Float16* arow = act + (size_t)(m0 + mloc) * (2 * L1);
  for (int ks = 0; ks < 16; ++ks) {
    const _Float16* ab = arow + ks * 32;
    union { v16h v; v8h h[2]; } A;
    A.h[0] = *(const v8h*)(ab + half * 16);
    A.h[1] = *(const v8h*)(ab + half * 16 + 8);
    v16h B0 = *(const v16h*)(W1f + ((size_t)(ks * 2 + 0) * 32 + lane) * 16);
    v16h B1 = *(const v16h*)(W1f + ((size_t)(ks * 2 + 1) * 32 + lane) * 16);
    c0 = __builtin_amdgcn_wmma_f32_16x16x32_f16(false, A.v, false, B0, (short)0, c0, false, false);
    c1 = __builtin_amdgcn_wmma_f32_16x16x32_f16(false, A.v, false, B1, (short)0, c1, false, false);
  }
  // bias + clipped ReLU, reshuffle C-layout -> row-major f16 in LDS
#pragma unroll
  for (int r = 0; r < 8; ++r) {
    int m = r + 8 * half;
    float a = fminf(fmaxf(c0[r] + b1[n],      0.f), 1.f);
    float b = fminf(fmaxf(c1[r] + b1[16 + n], 0.f), 1.f);
    t1[m * 32 + n]      = (_Float16)a;
    t1[m * 32 + 16 + n] = (_Float16)b;
  }
  __syncthreads();

  // ---- Layer 2: [16x32] @ W2[32x32] ----
  union { v16h v; v8h h[2]; } A2;
  A2.h[0] = *(const v8h*)(t1 + mloc * 32 + half * 16);
  A2.h[1] = *(const v8h*)(t1 + mloc * 32 + half * 16 + 8);
  v16h B20 = *(const v16h*)(W2f + ((size_t)0 * 32 + lane) * 16);
  v16h B21 = *(const v16h*)(W2f + ((size_t)1 * 32 + lane) * 16);
  v8f d0 = {}, d1 = {};
  d0 = __builtin_amdgcn_wmma_f32_16x16x32_f16(false, A2.v, false, B20, (short)0, d0, false, false);
  d1 = __builtin_amdgcn_wmma_f32_16x16x32_f16(false, A2.v, false, B21, (short)0, d1, false, false);
#pragma unroll
  for (int r = 0; r < 8; ++r) {
    int m = r + 8 * half;
    t2[m * 32 + n]      = fminf(fmaxf(d0[r] + b2[n],      0.f), 1.f);
    t2[m * 32 + 16 + n] = fminf(fmaxf(d1[r] + b2[16 + n], 0.f), 1.f);
  }
  __syncthreads();

  // ---- Output layer: [16x32] . Wo + bo ----
  if (lane < 16) {
    float s = bo[0];
#pragma unroll
    for (int k = 0; k < 32; ++k) s += t2[lane * 32 + k] * Wo[k];
    out[m0 + lane] = s;
  }
}

// ---------------------------------------------------------------------------
extern "C" void kernel_launch(void* const* d_in, const int* in_sizes, int n_in,
                              void* d_out, int out_size, void* d_ws, size_t ws_size,
                              hipStream_t stream) {
  const float* x1  = (const float*)d_in[0];
  const float* x2  = (const float*)d_in[1];
  const float* Win = (const float*)d_in[2];
  const float* bin = (const float*)d_in[3];
  const float* W1  = (const float*)d_in[4];
  const float* b1  = (const float*)d_in[5];
  const float* W2  = (const float*)d_in[6];
  const float* b2  = (const float*)d_in[7];
  const float* Wo  = (const float*)d_in[8];
  const float* bo  = (const float*)d_in[9];
  float* out = (float*)d_out;

  char* ws = (char*)d_ws;
  _Float16* act = (_Float16*)ws;                           // 1024*512*2 = 1 MiB
  _Float16* W1f = (_Float16*)(ws + (1u << 20));            // 32 KiB
  _Float16* W2f = (_Float16*)(ws + (1u << 20) + 32768);    // 2 KiB

  repack_weights<<<34, 256, 0, stream>>>(W1, W2, W1f, W2f);
  feature_transformer<<<2 * BATCH, 256, 0, stream>>>(x1, x2, Win, bin, act);
  mlp_head<<<BATCH / 16, 32, 0, stream>>>(act, W1f, b1, W2f, b2, Wo, bo, out);
}
